// CausalQSARModel_49529562857590
// MI455X (gfx1250) — matrix operand order repaired and checked by reading, not compiled, not run on hardware
//
#include <hip/hip_runtime.h>
#include <cstdint>
#include <cstddef>

// ---------------------------------------------------------------------------
// CausalQSAR GNN forward for gfx1250 (MI455X).
// GEMMs use V_WMMA_F32_16X16X32_BF16 (fp32 accum, bf16 inputs staged in LDS).
// Irregular graph ops (edge message/scatter, GraphNorm, pooling) use native
// fp32 global atomics.
// ---------------------------------------------------------------------------

typedef __attribute__((ext_vector_type(16))) __bf16 v16bf;
typedef __attribute__((ext_vector_type(8)))  float  v8f;

enum : int { NN = 100000, EE = 300000, GG = 4096,
             ND = 64, ED = 16, HH = 256, ZI = 128, ZS = 128, NEV = 8, LL = 3 };

#define DEV __device__ __forceinline__

DEV unsigned short f2bf(float f) {
  union { float f; uint32_t u; } c; c.f = f;
  uint32_t u = c.u;
  u += 0x7fffu + ((u >> 16) & 1u);   // round-to-nearest-even
  return (unsigned short)(u >> 16);
}

DEV void atomAddF(float* p, float v) {
  // lowers to global_atomic_add_f32 (no CAS loop)
  unsafeAtomicAdd(p, v);
}

// ---------------------------------------------------------------------------
// WMMA GEMM: C[M,Nc] = act( (alpha*A + A2) @ W + bias ), W row-major [K,Nc].
// Block = 256 threads (8 waves), tile 128x128, wave tile 32x64 (2x4 WMMA).
// A staged fp32->bf16 row-major in LDS; W staged transposed [n][k] so both
// operand fragments are contiguous 16B ds_load_b128 per the CDNA5 16-bit
// A/B VGPR layouts (lane&15 = row, lane>>4 selects the K half).
// Requires K % 32 == 0 (true for all uses: 64/128/256).
// ---------------------------------------------------------------------------
#define BM 128
#define BN 128
#define BK 32
#define LSTR 40   // ushorts per LDS row (32 + 8 pad) = 80B, 16B-aligned rows

template<bool RELU>
__global__ __launch_bounds__(256)
void wmma_gemm(const float* __restrict__ A, const float* __restrict__ A2,
               const float* __restrict__ epsPtr,
               const float* __restrict__ W, const float* __restrict__ bias,
               float* __restrict__ C, int M, int Nc, int K)
{
  __shared__ unsigned short sA[BM * LSTR];
  __shared__ unsigned short sB[BN * LSTR];

  const int tid  = threadIdx.x;
  const int wid  = tid >> 5;
  const int lane = tid & 31;
  const int wm   = wid >> 1;        // 0..3 : 32-row slice
  const int wn   = wid & 1;         // 0..1 : 64-col slice
  const int mrow = lane & 15;
  const int sel  = lane >> 4;       // 0/1 : K-half per ISA layout

  const int rowBase = blockIdx.y * BM;
  const int colBase = blockIdx.x * BN;
  const float alpha = epsPtr ? (1.0f + *epsPtr) : 1.0f;

  v8f acc[2][4];
#pragma unroll
  for (int mi = 0; mi < 2; ++mi)
#pragma unroll
    for (int ni = 0; ni < 4; ++ni)
#pragma unroll
      for (int r = 0; r < 8; ++r) acc[mi][ni][r] = 0.0f;

  for (int k0 = 0; k0 < K; k0 += BK) {
    // ---- stage A tile (128 x 32 fp32 -> bf16), fused alpha*A + A2 ----
    {
      const int r0 = tid >> 3;       // 0..31
      const int c4 = tid & 7;        // float4 column group
#pragma unroll
      for (int i = 0; i < 4; ++i) {
        const int row  = r0 + i * 32;
        const int grow = rowBase + row;
        float4 va = make_float4(0.f, 0.f, 0.f, 0.f);
        if (grow < M) {
          va = *reinterpret_cast<const float4*>(A + (size_t)grow * K + k0 + c4 * 4);
          if (A2) {
            float4 vb = *reinterpret_cast<const float4*>(A2 + (size_t)grow * K + k0 + c4 * 4);
            va.x = alpha * va.x + vb.x; va.y = alpha * va.y + vb.y;
            va.z = alpha * va.z + vb.z; va.w = alpha * va.w + vb.w;
          }
        }
        uint2 pk;
        pk.x = (uint32_t)f2bf(va.x) | ((uint32_t)f2bf(va.y) << 16);
        pk.y = (uint32_t)f2bf(va.z) | ((uint32_t)f2bf(va.w) << 16);
        *reinterpret_cast<uint2*>(&sA[row * LSTR + c4 * 4]) = pk;
      }
    }
    // ---- stage B tile transposed: sB[n][k] = W[k0+k][colBase+n] ----
    {
      const int kk = tid >> 3;       // 0..31
      const int n4 = tid & 7;
#pragma unroll
      for (int i = 0; i < 4; ++i) {
        const int nloc = (n4 + i * 8) * 4;
#pragma unroll
        for (int j = 0; j < 4; ++j) {
          const int gcol = colBase + nloc + j;
          const float w = (gcol < Nc) ? W[(size_t)(k0 + kk) * Nc + gcol] : 0.0f;
          sB[(nloc + j) * LSTR + kk] = f2bf(w);
        }
      }
    }
    __syncthreads();

    union FragU { uint4 q[2]; v16bf v; };
    FragU afrag[2], bfrag[4];
#pragma unroll
    for (int mi = 0; mi < 2; ++mi) {
      const unsigned short* base = &sA[(wm * 32 + mi * 16 + mrow) * LSTR];
      afrag[mi].q[0] = *reinterpret_cast<const uint4*>(base + sel * 8);        // K sel*8..+7
      afrag[mi].q[1] = *reinterpret_cast<const uint4*>(base + 16 + sel * 8);   // K 16+sel*8..+7
    }
#pragma unroll
    for (int ni = 0; ni < 4; ++ni) {
      const unsigned short* base = &sB[(wn * 64 + ni * 16 + mrow) * LSTR + sel * 8];
      bfrag[ni].q[0] = *reinterpret_cast<const uint4*>(base);                  // K sel*8..+7
      bfrag[ni].q[1] = *reinterpret_cast<const uint4*>(base + 16);             // K 16+sel*8..+7
    }
#pragma unroll
    for (int mi = 0; mi < 2; ++mi)
#pragma unroll
      for (int ni = 0; ni < 4; ++ni)
        acc[mi][ni] = __builtin_amdgcn_wmma_f32_16x16x32_bf16(
            false, afrag[mi].v, false, bfrag[ni].v, (short)0, acc[mi][ni], false, false);
    __syncthreads();
  }

  // ---- epilogue: bias (+ReLU), C layout: VGPR r -> row sel*8+r, col lane&15
#pragma unroll
  for (int mi = 0; mi < 2; ++mi) {
#pragma unroll
    for (int ni = 0; ni < 4; ++ni) {
      const int gcol = colBase + wn * 64 + ni * 16 + mrow;
      if (gcol >= Nc) continue;
      const float bv = bias ? bias[gcol] : 0.0f;
#pragma unroll
      for (int r = 0; r < 8; ++r) {
        const int grow = rowBase + wm * 32 + mi * 16 + sel * 8 + r;
        if (grow < M) {
          float v = acc[mi][ni][r] + bv;
          if (RELU) v = fmaxf(v, 0.0f);
          C[(size_t)grow * Nc + gcol] = v;
        }
      }
    }
  }
}

// ---------------------------------------------------------------------------
// Fused GINE edge op: per edge e, m = relu(h[src] + edge_attr@eW + eb),
// agg[dst] += m (atomic). eW/eb staged in LDS; wave handles one edge per
// iteration, each lane owns 8 contiguous channels (coalesced float4 loads).
// Avoids materializing the [E,256] message tensor (~600 MB of traffic saved).
// ---------------------------------------------------------------------------
__global__ __launch_bounds__(256)
void edge_msg(const float* __restrict__ h, const int* __restrict__ ei,
              const float* __restrict__ ea, const float* __restrict__ eW,
              const float* __restrict__ eb, float* __restrict__ agg, int E)
{
  __shared__ float sW[ED * HH];
  __shared__ float sb[HH];
  for (int i = threadIdx.x; i < ED * HH; i += 256) sW[i] = eW[i];
  if (threadIdx.x < HH) sb[threadIdx.x] = eb[threadIdx.x];
  __syncthreads();

  const int wid = threadIdx.x >> 5, lane = threadIdx.x & 31;
  const int c0 = lane * 8;
  for (int j = 0; j < 4; ++j) {
    const int e = blockIdx.x * 32 + wid * 4 + j;
    if (e >= E) break;
    const int src = ei[e], dst = ei[E + e];

    float eav[ED];
    const float4* eap = reinterpret_cast<const float4*>(ea + (size_t)e * ED);
#pragma unroll
    for (int q = 0; q < 4; ++q) {
      float4 t = eap[q];
      eav[q * 4 + 0] = t.x; eav[q * 4 + 1] = t.y;
      eav[q * 4 + 2] = t.z; eav[q * 4 + 3] = t.w;
    }
    const float4* hp = reinterpret_cast<const float4*>(h + (size_t)src * HH + c0);
    float4 h0 = hp[0], h1 = hp[1];
    float hv[8] = { h0.x, h0.y, h0.z, h0.w, h1.x, h1.y, h1.z, h1.w };

    float* ag = agg + (size_t)dst * HH + c0;
#pragma unroll
    for (int c = 0; c < 8; ++c) {
      float accv = sb[c0 + c];
#pragma unroll
      for (int k = 0; k < ED; ++k) accv = fmaf(eav[k], sW[k * HH + c0 + c], accv);
      float m = fmaxf(hv[c] + accv, 0.0f);
      atomAddF(ag + c, m);
    }
  }
}

// ---------------------------------------------------------------------------
// Elementwise / segment kernels (H == 256 hardcoded: i>>8 = node, i&255 = ch)
// ---------------------------------------------------------------------------
__global__ void fill0(float* __restrict__ p, long long n) {
  long long i = (long long)blockIdx.x * 256 + threadIdx.x;
  if (i < n) p[i] = 0.0f;
}
__global__ void count_nodes(const int* __restrict__ batch, float* __restrict__ cnt, int n) {
  int i = blockIdx.x * 256 + threadIdx.x;
  if (i < n) atomAddF(&cnt[batch[i]], 1.0f);
}
__global__ void clamp1(float* __restrict__ cnt, int g) {
  int i = blockIdx.x * 256 + threadIdx.x;
  if (i < g) cnt[i] = fmaxf(cnt[i], 1.0f);
}
__global__ void seg_acc(const float* __restrict__ v, const int* __restrict__ batch,
                        float* __restrict__ out, long long n) {
  long long i = (long long)blockIdx.x * 256 + threadIdx.x;
  if (i < n) {
    int node = (int)(i >> 8), c = (int)(i & 255);
    atomAddF(&out[(size_t)batch[node] * HH + c], v[i]);
  }
}
__global__ void gn_center(float* __restrict__ h, const float* __restrict__ meanS,
                          const float* __restrict__ cnt, const int* __restrict__ batch,
                          const float* __restrict__ ms, float* __restrict__ varA, long long n) {
  long long i = (long long)blockIdx.x * 256 + threadIdx.x;
  if (i >= n) return;
  int node = (int)(i >> 8), c = (int)(i & 255);
  int b = batch[node];
  float mean = meanS[(size_t)b * HH + c] / cnt[b];
  float o = h[i] - ms[c] * mean;
  h[i] = o;
  atomAddF(&varA[(size_t)b * HH + c], o * o);
}
__global__ void gn_final(float* __restrict__ h, const float* __restrict__ varS,
                         const float* __restrict__ cnt, const int* __restrict__ batch,
                         const float* __restrict__ w, const float* __restrict__ bb, long long n) {
  long long i = (long long)blockIdx.x * 256 + threadIdx.x;
  if (i >= n) return;
  int node = (int)(i >> 8), c = (int)(i & 255);
  int b = batch[node];
  float var = varS[(size_t)b * HH + c] / cnt[b];
  float o = h[i] * rsqrtf(var + 1e-5f) * w[c] + bb[c];
  h[i] = fmaxf(o, 0.0f);
}
__global__ void pool_div(float* __restrict__ hg, const float* __restrict__ cnt, long long n) {
  long long i = (long long)blockIdx.x * 256 + threadIdx.x;
  if (i < n) hg[i] /= cnt[i >> 8];
}

// ---------------------------------------------------------------------------
extern "C" void kernel_launch(void* const* d_in, const int* in_sizes, int n_in,
                              void* d_out, int out_size, void* d_ws, size_t ws_size,
                              hipStream_t stream)
{
  const float* x       = (const float*)d_in[0];
  const int*   eidx    = (const int*)  d_in[1];
  const float* eattr   = (const float*)d_in[2];
  const int*   batch   = (const int*)  d_in[3];
  const float* node_W  = (const float*)d_in[4];
  const float* node_b  = (const float*)d_in[5];
  const float* eps     = (const float*)d_in[6];
  const float* edge_W  = (const float*)d_in[7];
  const float* edge_b  = (const float*)d_in[8];
  const float* mlp_W1  = (const float*)d_in[9];
  const float* mlp_b1  = (const float*)d_in[10];
  const float* mlp_W2  = (const float*)d_in[11];
  const float* mlp_b2  = (const float*)d_in[12];
  const float* gn_w    = (const float*)d_in[13];
  const float* gn_b    = (const float*)d_in[14];
  const float* gn_ms   = (const float*)d_in[15];
  const float* finv_W1 = (const float*)d_in[16];
  const float* finv_b1 = (const float*)d_in[17];
  const float* finv_W2 = (const float*)d_in[18];
  const float* finv_b2 = (const float*)d_in[19];
  const float* fspu_W1 = (const float*)d_in[20];
  const float* fspu_b1 = (const float*)d_in[21];
  const float* fspu_W2 = (const float*)d_in[22];
  const float* fspu_b2 = (const float*)d_in[23];
  const float* pred_W1 = (const float*)d_in[24];
  const float* pred_b1 = (const float*)d_in[25];
  const float* pred_W2 = (const float*)d_in[26];
  const float* pred_b2 = (const float*)d_in[27];
  const float* adv_W1  = (const float*)d_in[28];
  const float* adv_b1  = (const float*)d_in[29];
  const float* adv_W2  = (const float*)d_in[30];
  const float* adv_b2  = (const float*)d_in[31];

  float* out   = (float*)d_out;
  float* hg    = out;                              // [G,H]
  float* z_inv = hg    + (size_t)GG * HH;          // [G,ZI]
  float* z_spu = z_inv + (size_t)GG * ZI;          // [G,ZS]
  float* yhat  = z_spu + (size_t)GG * ZS;          // [G]
  float* envh  = yhat  + GG;                       // [G,NE]

  const size_t NH = (size_t)NN * HH, GH = (size_t)GG * HH;
  float* ws    = (float*)d_ws;
  float* h     = ws;                               // [N,H]
  float* agg   = h     + NH;                       // [N,H]
  float* t1    = agg   + NH;                       // [N,H]
  float* meanb = t1    + NH;                       // [G,H]
  float* varb  = meanb + GH;                       // [G,H]
  float* cnt   = varb  + GH;                       // [G]
  float* th    = cnt   + GG;                       // [G,ZI] head temp

  auto g1 = [](long long n) { return dim3((unsigned)((n + 255) / 256)); };
  auto gg = [](int M, int Ncol) { return dim3((Ncol + 127) / 128, (M + 127) / 128); };
  const dim3 blk(256);

  // node counts per graph (cnt = max(segment_sum(1), 1))
  fill0<<<g1(GG), blk, 0, stream>>>(cnt, GG);
  count_nodes<<<g1(NN), blk, 0, stream>>>(batch, cnt, NN);
  clamp1<<<g1(GG), blk, 0, stream>>>(cnt, GG);

  // h = x @ node_W + node_b
  wmma_gemm<false><<<gg(NN, HH), blk, 0, stream>>>(x, nullptr, nullptr, node_W, node_b, h, NN, HH, ND);

  for (int l = 0; l < LL; ++l) {
    // agg = segment_sum_dst( relu(h[src] + edge_attr@eW + eb) )
    fill0<<<g1((long long)NH), blk, 0, stream>>>(agg, (long long)NH);
    edge_msg<<<dim3((EE + 31) / 32), blk, 0, stream>>>(
        h, eidx, eattr, edge_W + (size_t)l * ED * HH, edge_b + (size_t)l * HH, agg, EE);
    // t1 = relu(((1+eps)*h + agg) @ W1 + b1) ; h = t1 @ W2 + b2
    wmma_gemm<true ><<<gg(NN, HH), blk, 0, stream>>>(h, agg, eps + l,
        mlp_W1 + (size_t)l * HH * HH, mlp_b1 + (size_t)l * HH, t1, NN, HH, HH);
    wmma_gemm<false><<<gg(NN, HH), blk, 0, stream>>>(t1, nullptr, nullptr,
        mlp_W2 + (size_t)l * HH * HH, mlp_b2 + (size_t)l * HH, h, NN, HH, HH);
    // GraphNorm + ReLU
    fill0<<<g1((long long)GH), blk, 0, stream>>>(meanb, (long long)GH);
    fill0<<<g1((long long)GH), blk, 0, stream>>>(varb, (long long)GH);
    seg_acc<<<g1((long long)NH), blk, 0, stream>>>(h, batch, meanb, (long long)NH);
    gn_center<<<g1((long long)NH), blk, 0, stream>>>(h, meanb, cnt, batch,
        gn_ms + (size_t)l * HH, varb, (long long)NH);
    gn_final<<<g1((long long)NH), blk, 0, stream>>>(h, varb, cnt, batch,
        gn_w + (size_t)l * HH, gn_b + (size_t)l * HH, (long long)NH);
  }

  // global mean pool -> hg (in d_out)
  fill0<<<g1((long long)GH), blk, 0, stream>>>(hg, (long long)GH);
  seg_acc<<<g1((long long)NH), blk, 0, stream>>>(h, batch, hg, (long long)NH);
  pool_div<<<g1((long long)GH), blk, 0, stream>>>(hg, cnt, (long long)GH);

  // heads (grl == z_inv numerically in forward)
  wmma_gemm<true ><<<gg(GG, ZI), blk, 0, stream>>>(hg, nullptr, nullptr, finv_W1, finv_b1, th, GG, ZI, HH);
  wmma_gemm<false><<<gg(GG, ZI), blk, 0, stream>>>(th, nullptr, nullptr, finv_W2, finv_b2, z_inv, GG, ZI, ZI);
  wmma_gemm<true ><<<gg(GG, ZS), blk, 0, stream>>>(hg, nullptr, nullptr, fspu_W1, fspu_b1, th, GG, ZS, HH);
  wmma_gemm<false><<<gg(GG, ZS), blk, 0, stream>>>(th, nullptr, nullptr, fspu_W2, fspu_b2, z_spu, GG, ZS, ZS);
  wmma_gemm<true ><<<gg(GG, ZI), blk, 0, stream>>>(z_inv, nullptr, nullptr, pred_W1, pred_b1, th, GG, ZI, ZI);
  wmma_gemm<false><<<gg(GG, 1 ), blk, 0, stream>>>(th, nullptr, nullptr, pred_W2, pred_b2, yhat, GG, 1, ZI);
  wmma_gemm<true ><<<gg(GG, ZI), blk, 0, stream>>>(z_inv, nullptr, nullptr, adv_W1, adv_b1, th, GG, ZI, ZI);
  wmma_gemm<false><<<gg(GG, NEV), blk, 0, stream>>>(th, nullptr, nullptr, adv_W2, adv_b2, envh, GG, NEV, ZI);
}
